// DeadRecsGNN_68083821576206
// MI455X (gfx1250) — matrix operand order repaired and checked by reading, not compiled, not run on hardware
//
#include <hip/hip_runtime.h>
#include <hip/hip_bf16.h>

// ---------------------------------------------------------------------------
// Hetero GraphSAGE (2 layers, D=256) for MI455X / gfx1250.
//  * segment-mean via global f32 atomics (HBM-bound phase)
//  * all GEMMs via v_wmma_f32_16x16x32_bf16 (f32 accumulate)
//  * A tiles double-buffered via global_load_async_to_lds_b128 (ASYNCcnt),
//    DMA of tile k+1 overlaps WMMA of tile k; ONE barrier per K step
//  * whole 64-col W panel staged to LDS once (transposed bf16)
//  * W_r / b_l summed per destination type (algebraic folding)
//  * bias / accumulate / relu fused into the GEMM epilogue
// Workspace (f32): mean buffer N_max*256 + counts + layer-1 activations
// (sum N)*256 + 3*256*256 + 3*256  ->  ~360 MB for the reference sizes.
// ---------------------------------------------------------------------------

typedef __attribute__((ext_vector_type(16))) __bf16 v16bf;
typedef __attribute__((ext_vector_type(8)))  __bf16 v8bf;
typedef __attribute__((ext_vector_type(8)))  float  v8f;
typedef __attribute__((ext_vector_type(4)))  float  v4f;

#define BM 128
#define BN 64
#define BK 32
#define ASTF 36   // LDS row stride (f32) for A tile: 36*4 = 144 B (16B aligned)
#define WST  264  // LDS row stride (halves) for W panel: 264*2 = 528 B (16B aligned)

// ------------------------------ utility kernels ----------------------------

__global__ void zero_kernel(float* __restrict__ p, size_t n) {
  size_t i = (size_t)blockIdx.x * blockDim.x + threadIdx.x;
  size_t stride = (size_t)gridDim.x * blockDim.x;
  for (; i < n; i += stride) p[i] = 0.0f;
}

// out[i] = W[k0][i] + W[k1][i] (+ W[k2][i] if nk==3); W is [7][256*256]
__global__ void wsum_kernel(const float* __restrict__ W, float* __restrict__ out,
                            int nk, int k0, int k1, int k2) {
  int i = blockIdx.x * 256 + threadIdx.x;
  if (i >= 65536) return;
  float s = W[(size_t)k0 * 65536 + i] + W[(size_t)k1 * 65536 + i];
  if (nk == 3) s += W[(size_t)k2 * 65536 + i];
  out[i] = s;
}

// bias fold: b is [7][256]
__global__ void bsum_kernel(const float* __restrict__ b, float* __restrict__ out,
                            int nk, int k0, int k1, int k2) {
  int i = threadIdx.x;
  float s = b[k0 * 256 + i] + b[k1 * 256 + i];
  if (nk == 3) s += b[k2 * 256 + i];
  out[i] = s;
}

// one block per edge: acc[dst] += x[src] (256 dims), cnt[dst] += 1
__global__ void scatter_add_kernel(const float* __restrict__ x,
                                   const int* __restrict__ src,
                                   const int* __restrict__ dst,
                                   float* __restrict__ acc,
                                   float* __restrict__ cnt, int E) {
  int e = blockIdx.x;
  if (e >= E) return;
  int s = src[e], d = dst[e];
  const float* xs = x + (size_t)s * 256;
  float* ad = acc + (size_t)d * 256;
  atomicAdd(&ad[threadIdx.x], xs[threadIdx.x]);
  if (threadIdx.x == 0) atomicAdd(&cnt[d], 1.0f);
}

// acc[i] /= max(cnt[row],1)
__global__ void mean_div_kernel(float* __restrict__ acc,
                                const float* __restrict__ cnt, int M) {
  size_t i = (size_t)blockIdx.x * 256 + threadIdx.x;
  if (i >= (size_t)M * 256) return;
  float c = cnt[i >> 8];
  acc[i] *= (1.0f / (c < 1.0f ? 1.0f : c));
}

// ------------------------------ WMMA GEMM ----------------------------------
// C[M,256] (flags&1: +=) A[M,256] @ W[256,256]  (+ bias[256] if flags&2),
// relu on store if flags&4.  A,W are f32 in memory.
// Block: 256 threads (8 waves). Block tile 128x64; wave tile 16x64 (4 WMMA).
// Out-of-range A rows are CLAMPED (not zeroed): they only feed C rows that
// are never stored, so no divergent guards exist in the K loop and EXEC
// stays all-1s (as WMMA requires).
// Pipeline per K step (one barrier): wait asynccnt 0 -> barrier ->
// issue async copy of tile k+1 -> WMMA on tile k.
__global__ __launch_bounds__(256)
void sage_gemm_wmma(const float* __restrict__ A, const float* __restrict__ W,
                    const float* __restrict__ bias, float* __restrict__ C,
                    int M, int flags) {
  __shared__ __align__(16) float  As[2][BM * ASTF];  // double-buffered A tile
  __shared__ __align__(16) __bf16 Ws[BN * WST];      // full W panel, transposed

  const int tid  = threadIdx.x;
  const int wv   = tid >> 5;        // wave 0..7 -> 16-row strip
  const int lane = tid & 31;
  const int l16  = lane & 15;
  const int hi   = lane >> 4;       // half-wave select
  const int m0   = blockIdx.x * BM;
  const int n0   = blockIdx.y * BN;

  v8f acc[4];
#pragma unroll
  for (int j = 0; j < 4; ++j) {
    int gn = n0 + j * 16 + l16;
    float bv = (flags & 2) ? bias[gn] : 0.0f;
#pragma unroll
    for (int r = 0; r < 8; ++r) {
      int gm = m0 + wv * 16 + hi * 8 + r;
      float cv = 0.0f;
      if ((flags & 1) && gm < M) cv = C[(size_t)gm * 256 + gn];
      acc[j][r] = cv + bv;
    }
  }

  // this thread's A-tile async-copy slots (K-invariant)
  int arow[4], acol[4];
  const float* abase[4];
#pragma unroll
  for (int i = 0; i < 4; ++i) {
    int idx = tid + 256 * i;            // 0..1023 16-byte slots
    arow[i] = idx >> 3;
    acol[i] = (idx & 7) * 4;
    int gm = m0 + arow[i];
    if (gm > M - 1) gm = M - 1;         // clamp: garbage rows never stored
    abase[i] = A + (size_t)gm * 256 + acol[i];
  }

  // ---- prologue: kick off async copy of A tile 0 ----
#pragma unroll
  for (int i = 0; i < 4; ++i) {
    unsigned lds_addr = (unsigned)(size_t)(&As[0][arow[i] * ASTF + acol[i]]);
    asm volatile("global_load_async_to_lds_b128 %0, %1, off"
                 :: "v"(lds_addr), "v"(abase[i]) : "memory");
  }

  // ---- stage full W panel once: Ws[col][k], transposed bf16 ----
#pragma unroll
  for (int i = 0; i < 16; ++i) {
    int idx = tid + 256 * i;            // 0..4095 float4 slots (256 k x 16 col4)
    int k   = idx >> 4;
    int c4  = (idx & 15) * 4;
    float4 v = *reinterpret_cast<const float4*>(W + (size_t)k * 256 + n0 + c4);
    Ws[(c4 + 0) * WST + k] = (__bf16)v.x;
    Ws[(c4 + 1) * WST + k] = (__bf16)v.y;
    Ws[(c4 + 2) * WST + k] = (__bf16)v.z;
    Ws[(c4 + 3) * WST + k] = (__bf16)v.w;
  }

  for (int it = 0; it < 8; ++it) {
    const int kk = it * BK;
    // tile `it` DMA complete (overlapped with previous iteration's WMMAs)
    asm volatile("s_wait_asynccnt 0x0" ::: "memory");
    // all waves' tile-`it` writes visible; all reads of the other buffer done
    __syncthreads();
    // launch DMA for tile it+1 into the other buffer; overlaps WMMAs below
    if (it < 7) {
#pragma unroll
      for (int i = 0; i < 4; ++i) {
        unsigned lds_addr =
            (unsigned)(size_t)(&As[(it + 1) & 1][arow[i] * ASTF + acol[i]]);
        const float* gp = abase[i] + (kk + BK);
        asm volatile("global_load_async_to_lds_b128 %0, %1, off"
                     :: "v"(lds_addr), "v"(gp) : "memory");
      }
    }

    // ---- A fragment (16x32 bf16): lane = row; K halves {klo..+7, klo+16..+23}
    const float* ar = &As[it & 1][(wv * 16 + l16) * ASTF];
    int klo = hi * 8;
    v4f a0 = *reinterpret_cast<const v4f*>(ar + klo);
    v4f a1 = *reinterpret_cast<const v4f*>(ar + klo + 4);
    v4f a2 = *reinterpret_cast<const v4f*>(ar + klo + 16);
    v4f a3 = *reinterpret_cast<const v4f*>(ar + klo + 20);
    v16bf afrag;
#pragma unroll
    for (int e = 0; e < 4; ++e) {
      afrag[e]      = (__bf16)a0[e];
      afrag[e + 4]  = (__bf16)a1[e];
      afrag[e + 8]  = (__bf16)a2[e];
      afrag[e + 12] = (__bf16)a3[e];
    }

#pragma unroll
    for (int j = 0; j < 4; ++j) {
      // ---- B fragment (32x16 bf16): lane = col; K halves {klo..+7, klo+16..+23}
      //      (same k order as A)
      const __bf16* br = &Ws[(j * 16 + l16) * WST + kk + hi * 8];
      v8bf b_lo = *reinterpret_cast<const v8bf*>(br);
      v8bf b_hi = *reinterpret_cast<const v8bf*>(br + 16);
      v16bf bfrag;
#pragma unroll
      for (int e = 0; e < 8; ++e) { bfrag[e] = b_lo[e]; bfrag[e + 8] = b_hi[e]; }

      acc[j] = __builtin_amdgcn_wmma_f32_16x16x32_bf16(
          /*neg_a=*/false, afrag, /*neg_b=*/false, bfrag,
          /*c_mod=*/(short)0, acc[j], /*reuse_a=*/false, /*reuse_b=*/false);
    }
  }

  // ---- epilogue ----
#pragma unroll
  for (int j = 0; j < 4; ++j) {
    int gn = n0 + j * 16 + l16;
#pragma unroll
    for (int r = 0; r < 8; ++r) {
      int gm = m0 + wv * 16 + hi * 8 + r;
      if (gm < M) {
        float v = acc[j][r];
        if (flags & 4) v = fmaxf(v, 0.0f);
        C[(size_t)gm * 256 + gn] = v;
      }
    }
  }
}

// ------------------------------ launcher -----------------------------------

extern "C" void kernel_launch(void* const* d_in, const int* in_sizes, int n_in,
                              void* d_out, int out_size, void* d_ws, size_t ws_size,
                              hipStream_t stream) {
  const float* X[3]  = {(const float*)d_in[0], (const float*)d_in[1], (const float*)d_in[2]};
  int N[3] = {in_sizes[0] / 256, in_sizes[1] / 256, in_sizes[2] / 256};
  const float* Wl[2] = {(const float*)d_in[3], (const float*)d_in[6]};
  const float* Bl[2] = {(const float*)d_in[4], (const float*)d_in[7]};
  const float* Wr[2] = {(const float*)d_in[5], (const float*)d_in[8]};
  const int* EI[7]; int E[7];
  for (int k = 0; k < 7; ++k) { EI[k] = (const int*)d_in[9 + k]; E[k] = in_sizes[9 + k] / 2; }
  // EDGE_DEFS: src / dst type per edge type (0=show,1=perf,2=song)
  const int esrc[7] = {0, 1, 2, 0, 1, 2, 2};
  const int edst[7] = {1, 2, 2, 0, 0, 1, 2};

  size_t NMAX = (size_t)N[0];
  if ((size_t)N[1] > NMAX) NMAX = N[1];
  if ((size_t)N[2] > NMAX) NMAX = N[2];
  size_t Ntot = (size_t)N[0] + N[1] + N[2];

  float* meanbuf = (float*)d_ws;                       // NMAX*256
  float* cntbuf  = meanbuf + NMAX * 256;               // NMAX (padded)
  float* Hbuf    = cntbuf + ((NMAX + 255) & ~(size_t)255);  // Ntot*256 (layer-1 act)
  float* wr3     = Hbuf + Ntot * 256;                  // 3*65536
  float* bs3     = wr3 + 3 * 65536;                    // 3*256

  float* Hp[3] = {Hbuf, Hbuf + (size_t)N[0] * 256, Hbuf + ((size_t)N[0] + N[1]) * 256};
  float* Op[3] = {(float*)d_out, (float*)d_out + (size_t)N[0] * 256,
                  (float*)d_out + ((size_t)N[0] + N[1]) * 256};

  for (int l = 0; l < 2; ++l) {
    const float* Xin[3]; float* Yout[3];
    for (int t = 0; t < 3; ++t) {
      Xin[t]  = (l == 0) ? X[t] : (const float*)Hp[t];
      Yout[t] = (l == 0) ? Hp[t] : Op[t];
    }
    // fold W_r and b_l per destination type:
    // show <- k{3,4}; perf <- k{0,5}; song <- k{1,2,6}
    wsum_kernel<<<256, 256, 0, stream>>>(Wr[l], wr3 + 0 * 65536, 2, 3, 4, 0);
    wsum_kernel<<<256, 256, 0, stream>>>(Wr[l], wr3 + 1 * 65536, 2, 0, 5, 0);
    wsum_kernel<<<256, 256, 0, stream>>>(Wr[l], wr3 + 2 * 65536, 3, 1, 2, 6);
    bsum_kernel<<<1, 256, 0, stream>>>(Bl[l], bs3 + 0,   2, 3, 4, 0);
    bsum_kernel<<<1, 256, 0, stream>>>(Bl[l], bs3 + 256, 2, 0, 5, 0);
    bsum_kernel<<<1, 256, 0, stream>>>(Bl[l], bs3 + 512, 3, 1, 2, 6);

    // self term: Y[t] = X[t] @ (sum W_r) + (sum b_l)   [flags: bias]
    for (int t = 0; t < 3; ++t) {
      dim3 g((N[t] + BM - 1) / BM, 256 / BN);
      sage_gemm_wmma<<<g, 256, 0, stream>>>(Xin[t], wr3 + (size_t)t * 65536,
                                            bs3 + t * 256, Yout[t], N[t], /*flags=*/2);
    }

    // edge terms, one edge type at a time (shared mean buffer)
    for (int k = 0; k < 7; ++k) {
      int st = esrc[k], dt = edst[k];
      size_t nm = (size_t)N[dt] * 256;
      zero_kernel<<<2048, 256, 0, stream>>>(meanbuf, nm);
      zero_kernel<<<256, 256, 0, stream>>>(cntbuf, (size_t)N[dt]);
      scatter_add_kernel<<<E[k], 256, 0, stream>>>(Xin[st], EI[k], EI[k] + E[k],
                                                   meanbuf, cntbuf, E[k]);
      mean_div_kernel<<<(unsigned)((nm + 255) / 256), 256, 0, stream>>>(meanbuf, cntbuf, N[dt]);

      int flags = 1;                                   // accumulate
      if (l == 0 && (k == 4 || k == 5 || k == 6)) flags |= 4;  // fuse ReLU on last term
      dim3 g((N[dt] + BM - 1) / BM, 256 / BN);
      sage_gemm_wmma<<<g, 256, 0, stream>>>(meanbuf, Wl[l] + (size_t)k * 65536,
                                            nullptr, Yout[dt], N[dt], flags);
    }
  }
}
